// SimplexLSTM_81140522156226
// MI455X (gfx1250) — matrix operand order repaired and checked by reading, not compiled, not run on hardware
//
#include <hip/hip_runtime.h>
#include <hip/hip_bf16.h>

// ---------------------------------------------------------------------------
// SimplexLSTM forward for MI455X (gfx1250, wave32, WMMA bf16)
//   VOCAB=50257, D=768, NL=12, B=2, L=512  -> M = B*L = 1024 rows
// ---------------------------------------------------------------------------

#define VOCAB 50257
#define DMODEL 768
#define NLAYER 12
#define BATCH 2
#define SEQ 512
#define MROWS (BATCH * SEQ)   // 1024

typedef __attribute__((ext_vector_type(16))) __bf16 v16bf;
typedef __attribute__((ext_vector_type(8)))  __bf16 v8bf;
typedef __attribute__((ext_vector_type(2)))  __bf16 v2bf;
typedef __attribute__((ext_vector_type(8)))  float  v8f;

__device__ __forceinline__ v16bf concat16(v8bf lo, v8bf hi) {
    return __builtin_shufflevector(lo, hi, 0, 1, 2, 3, 4, 5, 6, 7,
                                           8, 9, 10, 11, 12, 13, 14, 15);
}

// ---------------------------------------------------------------------------
// Embedding gather: x[row, d] = emb[tokens[row], d]
// ---------------------------------------------------------------------------
__global__ void embed_kernel(const int* __restrict__ tokens,
                             const float* __restrict__ emb,
                             float* __restrict__ x) {
    int idx = blockIdx.x * blockDim.x + threadIdx.x;
    const int total = MROWS * DMODEL;
    if (idx >= total) return;
    int row = idx / DMODEL;
    int d   = idx - row * DMODEL;
    x[idx] = emb[(size_t)tokens[row] * DMODEL + d];
}

// ---------------------------------------------------------------------------
// LayerNorm over last dim (768), output bf16 (WMMA A-operand precision)
// ---------------------------------------------------------------------------
__global__ void ln_kernel(const float* __restrict__ x,
                          const float* __restrict__ w,
                          const float* __restrict__ b,
                          __bf16* __restrict__ out) {
    __shared__ float red[256];
    const int row = blockIdx.x;
    const int tid = threadIdx.x;
    const float* xr = x + (size_t)row * DMODEL;

    float s = 0.f;
    for (int d = tid; d < DMODEL; d += 256) s += xr[d];
    red[tid] = s; __syncthreads();
    for (int off = 128; off > 0; off >>= 1) {
        if (tid < off) red[tid] += red[tid + off];
        __syncthreads();
    }
    float mean = red[0] * (1.0f / DMODEL);
    __syncthreads();

    float v = 0.f;
    for (int d = tid; d < DMODEL; d += 256) { float t = xr[d] - mean; v += t * t; }
    red[tid] = v; __syncthreads();
    for (int off = 128; off > 0; off >>= 1) {
        if (tid < off) red[tid] += red[tid + off];
        __syncthreads();
    }
    float rstd = rsqrtf(red[0] * (1.0f / DMODEL) + 1e-5f);

    __bf16* orow = out + (size_t)row * DMODEL;
    for (int d = tid; d < DMODEL; d += 256)
        orow[d] = (__bf16)((xr[d] - mean) * rstd * w[d] + b[d]);
}

// ---------------------------------------------------------------------------
// bf16 WMMA GEMM: dst = act(A*W + bias [+ resid])
//   block tile 128(M) x 128(N), K-step 32, 8 waves as 4(M) x 2(N),
//   wave tile 32 x 64 -> 2x4 accumulators = 8 WMMA / wave / K-step.
//   Double-buffered LDS, register-staged global loads, K+2 global prefetch.
// ACT: 0 none, 1 sigmoid, 2 tanh
// GUARDN=false: N % 128 == 0 and ldw % 4 == 0 (float4 weight loads, no guards)
// GUARDN=true : arbitrary N/ldw (scalar weight loads; UNIFORM full/partial
//               branch, per-element guards only in the single partial block)
// ---------------------------------------------------------------------------
template <int ACT, bool HAS_BIAS, bool HAS_RESID, bool GUARDN>
__global__ void __launch_bounds__(256)
gemm_bf16(const __bf16* __restrict__ A,
          const float* __restrict__ W, int ldw,
          const float* __restrict__ bias,
          const float* __restrict__ resid,
          float* __restrict__ dst,
          int N, int K) {
    __shared__ __bf16 As[2][128 * 32];   // [m][k]  row stride 32 (64 B)
    __shared__ __bf16 Bs[2][128 * 40];   // [n][k]  row stride 40 (80 B, 16B-aligned)

    const int tid    = threadIdx.x;
    const int lane   = tid & 31;
    const int waveId = tid >> 5;
    const int wm     = waveId >> 1;      // 0..3 (32-row group)
    const int wn     = waveId & 1;       // 0..1 (64-col group)
    const int half   = lane >> 4;        // 0/1
    const int l16    = lane & 15;

    const int blockM = blockIdx.x * 128;
    const int blockN = blockIdx.y * 128;
    const bool fullN = !GUARDN || (blockN + 128 <= N);   // uniform across block

    // ---- staging-register addressing ----
    const int ar  = tid >> 1;            // A: row 0..127
    const int seg = (tid & 1) << 4;      //    cols seg..seg+15
    const __bf16* aBase = A + (size_t)(blockM + ar) * K + seg;

    int bkk[2], bnv[2];
    const float* wBase[2];
#pragma unroll
    for (int rep = 0; rep < 2; ++rep) {
        int lin  = tid + rep * 256;          // 0..511 over (k-pair) x (n4)
        bkk[rep] = (lin >> 5) << 1;          // 0,2,...,30
        bnv[rep] = (lin & 31) << 2;          // 0,4,...,124
        wBase[rep] = W + (size_t)bkk[rep] * ldw + blockN + bnv[rep];
    }

    v8bf  aR0, aR1;
    float bRa[2][4], bRb[2][4];              // rows k, k+1

    auto load_regs = [&](int k0) {
        aR0 = *(const v8bf*)(aBase + k0);
        aR1 = *(const v8bf*)(aBase + k0 + 8);
#pragma unroll
        for (int rep = 0; rep < 2; ++rep) {
            const float* p0 = wBase[rep] + (size_t)k0 * ldw;
            const float* p1 = p0 + ldw;
            if constexpr (!GUARDN) {
                float4 f0 = *(const float4*)p0;
                float4 f1 = *(const float4*)p1;
                bRa[rep][0] = f0.x; bRa[rep][1] = f0.y; bRa[rep][2] = f0.z; bRa[rep][3] = f0.w;
                bRb[rep][0] = f1.x; bRb[rep][1] = f1.y; bRb[rep][2] = f1.z; bRb[rep][3] = f1.w;
            } else {
                if (fullN) {
#pragma unroll
                    for (int j = 0; j < 4; ++j) { bRa[rep][j] = p0[j]; bRb[rep][j] = p1[j]; }
                } else {
                    int gn = blockN + bnv[rep];
#pragma unroll
                    for (int j = 0; j < 4; ++j) {
                        bool ok = (gn + j) < N;
                        bRa[rep][j] = ok ? p0[j] : 0.f;
                        bRb[rep][j] = ok ? p1[j] : 0.f;
                    }
                }
            }
        }
    };
    auto store_regs = [&](int buf) {
        *(v8bf*)&As[buf][ar * 32 + seg]     = aR0;
        *(v8bf*)&As[buf][ar * 32 + seg + 8] = aR1;
#pragma unroll
        for (int rep = 0; rep < 2; ++rep) {
#pragma unroll
            for (int j = 0; j < 4; ++j) {
                v2bf p2 = { (__bf16)bRa[rep][j], (__bf16)bRb[rep][j] };  // k, k+1
                *(v2bf*)&Bs[buf][(bnv[rep] + j) * 40 + bkk[rep]] = p2;
            }
        }
    };

    v8f acc[2][4];
#pragma unroll
    for (int i = 0; i < 2; ++i)
#pragma unroll
        for (int j = 0; j < 4; ++j)
#pragma unroll
            for (int r = 0; r < 8; ++r) acc[i][j][r] = 0.f;

    const int NK = K >> 5;
    load_regs(0);
    store_regs(0);
    __syncthreads();

    for (int kt = 0; kt < NK; ++kt) {
        const int cur = kt & 1;
        if (kt + 1 < NK) {
            load_regs((kt + 1) << 5);
            if (kt + 2 < NK) {   // CDNA5 global_prefetch_b8 for the K+2 tile
                __builtin_prefetch(aBase + ((kt + 2) << 5), 0, 1);
                __builtin_prefetch(wBase[0] + (size_t)((kt + 2) << 5) * ldw, 0, 1);
            }
        }

        // ---- fragments per ISA VGPR layouts ----
        v16bf afrag[2], bfrag[4];
#pragma unroll
        for (int i = 0; i < 2; ++i) {
            const __bf16* arow = &As[cur][(wm * 32 + i * 16 + l16) * 32];
            v8bf lo = *(const v8bf*)(arow + (half << 3));        // K {0-7 | 8-15}
            v8bf hi = *(const v8bf*)(arow + 16 + (half << 3));   // K {16-23 | 24-31}
            afrag[i] = concat16(lo, hi);
        }
#pragma unroll
        for (int j = 0; j < 4; ++j) {
            const __bf16* bcol = &Bs[cur][(wn * 64 + j * 16 + l16) * 40 + (half << 3)];
            v8bf lo = *(const v8bf*)bcol;
            v8bf hi = *(const v8bf*)(bcol + 16);
            bfrag[j] = concat16(lo, hi);
        }
#pragma unroll
        for (int i = 0; i < 2; ++i)
#pragma unroll
            for (int j = 0; j < 4; ++j)
                acc[i][j] = __builtin_amdgcn_wmma_f32_16x16x32_bf16(
                    false, afrag[i], false, bfrag[j], (short)0, acc[i][j], false, false);

        if (kt + 1 < NK) store_regs(cur ^ 1);
        __syncthreads();
    }

    // ---- epilogue: bias / residual / activation, fp32 store ----
#pragma unroll
    for (int i = 0; i < 2; ++i) {
        const int mBase = blockM + wm * 32 + i * 16 + (half << 3);
#pragma unroll
        for (int j = 0; j < 4; ++j) {
            const int n = blockN + wn * 64 + j * 16 + l16;
            if (GUARDN && n >= N) continue;
            const float bv = HAS_BIAS ? bias[n] : 0.f;
#pragma unroll
            for (int r = 0; r < 8; ++r) {
                float v = acc[i][j][r] + bv;
                if (HAS_RESID) v += resid[(size_t)(mBase + r) * N + n];
                if (ACT == 1)      v = 1.f / (1.f + __expf(-v));
                else if (ACT == 2) v = tanhf(v);
                dst[(size_t)(mBase + r) * N + n] = v;
            }
        }
    }
}

// ---------------------------------------------------------------------------
// Sequential gated scan over time:
//   c_t = f_t * c_{t-1} + exp(eg*i_t) * g_t ;  h_t = o_t * tanh(c_t)
// one lane per (batch, channel), coalesced over channel dim
// ---------------------------------------------------------------------------
__global__ void scan_kernel(const float* __restrict__ gi,
                            const float* __restrict__ gf,
                            const float* __restrict__ gg,
                            const float* __restrict__ go,
                            const float* __restrict__ eg,
                            __bf16* __restrict__ h) {
    int idx = blockIdx.x * blockDim.x + threadIdx.x;
    if (idx >= BATCH * DMODEL) return;
    int b = idx / DMODEL;
    int d = idx - b * DMODEL;
    float e = eg[d];
    float c = 0.f;
    for (int t = 0; t < SEQ; ++t) {
        size_t o = ((size_t)(b * SEQ + t)) * DMODEL + d;
        c = gf[o] * c + __expf(e * gi[o]) * gg[o];
        h[o] = (__bf16)(go[o] * tanhf(c));
    }
}

// ---------------------------------------------------------------------------
// Host-side launcher
// ---------------------------------------------------------------------------
extern "C" void kernel_launch(void* const* d_in, const int* in_sizes, int n_in,
                              void* d_out, int out_size, void* d_ws, size_t ws_size,
                              hipStream_t stream) {
    (void)in_sizes; (void)n_in; (void)out_size; (void)ws_size;

    const int*   tokens  = (const int*)  d_in[0];
    const float* emb     = (const float*)d_in[1];
    const float* norm_w  = (const float*)d_in[2];
    const float* norm_b  = (const float*)d_in[3];
    const float* ip_w    = (const float*)d_in[4];
    const float* ip_b    = (const float*)d_in[5];
    const float* fp_w    = (const float*)d_in[6];
    const float* fp_b    = (const float*)d_in[7];
    const float* cp_w    = (const float*)d_in[8];
    const float* cp_b    = (const float*)d_in[9];
    const float* op_w    = (const float*)d_in[10];
    const float* op_b    = (const float*)d_in[11];
    const float* expgate = (const float*)d_in[12];
    const float* outp_w  = (const float*)d_in[13];
    const float* lnf_w   = (const float*)d_in[14];
    const float* lnf_b   = (const float*)d_in[15];
    const float* head_w  = (const float*)d_in[16];
    float*       out     = (float*)d_out;

    const size_t mrows = MROWS, dm = DMODEL;
    char* p = (char*)d_ws;
    float*  x    = (float*)p;   p += mrows * dm * sizeof(float);
    __bf16* xnb  = (__bf16*)p;  p += mrows * dm * sizeof(__bf16);
    float*  gi   = (float*)p;   p += mrows * dm * sizeof(float);
    float*  gf   = (float*)p;   p += mrows * dm * sizeof(float);
    float*  gg   = (float*)p;   p += mrows * dm * sizeof(float);
    float*  go   = (float*)p;   p += mrows * dm * sizeof(float);
    __bf16* hb   = (__bf16*)p;  p += mrows * dm * sizeof(__bf16);

    {
        int total = MROWS * DMODEL;
        embed_kernel<<<(total + 255) / 256, 256, 0, stream>>>(tokens, emb, x);
    }

    const dim3 grid_d(MROWS / 128, DMODEL / 128);            // 8 x 6
    const dim3 grid_h(MROWS / 128, (VOCAB + 127) / 128);     // 8 x 393

    for (int l = 0; l < NLAYER; ++l) {
        const float* nw  = norm_w + (size_t)l * DMODEL;
        const float* nb  = norm_b + (size_t)l * DMODEL;
        const float* ipw = ip_w   + (size_t)l * DMODEL * 4 * DMODEL;
        const float* ipb = ip_b   + (size_t)l * 4 * DMODEL;
        const float* fpw = fp_w   + (size_t)l * DMODEL * DMODEL;
        const float* fpb = fp_b   + (size_t)l * DMODEL;
        const float* cpw = cp_w   + (size_t)l * DMODEL * DMODEL;
        const float* cpb = cp_b   + (size_t)l * DMODEL;
        const float* opw = op_w   + (size_t)l * DMODEL * DMODEL;
        const float* opb = op_b   + (size_t)l * DMODEL;
        const float* eg  = expgate+ (size_t)l * DMODEL;
        const float* ow  = outp_w + (size_t)l * DMODEL * DMODEL;

        ln_kernel<<<MROWS, 256, 0, stream>>>(x, nw, nb, xnb);

        gemm_bf16<1, true, false, false><<<grid_d, 256, 0, stream>>>(
            xnb, ipw, 4 * DMODEL, ipb, nullptr, gi, DMODEL, DMODEL);
        gemm_bf16<1, true, false, false><<<grid_d, 256, 0, stream>>>(
            xnb, fpw, DMODEL, fpb, nullptr, gf, DMODEL, DMODEL);
        gemm_bf16<2, true, false, false><<<grid_d, 256, 0, stream>>>(
            xnb, cpw, DMODEL, cpb, nullptr, gg, DMODEL, DMODEL);
        gemm_bf16<1, true, false, false><<<grid_d, 256, 0, stream>>>(
            xnb, opw, DMODEL, opb, nullptr, go, DMODEL, DMODEL);

        scan_kernel<<<(BATCH * DMODEL + 255) / 256, 256, 0, stream>>>(
            gi, gf, gg, go, eg, hb);

        gemm_bf16<0, false, true, false><<<grid_d, 256, 0, stream>>>(
            hb, ow, DMODEL, nullptr, x, x, DMODEL, DMODEL);
    }

    ln_kernel<<<MROWS, 256, 0, stream>>>(x, lnf_w, lnf_b, xnb);
    gemm_bf16<0, false, false, true><<<grid_h, 256, 0, stream>>>(
        xnb, head_w, VOCAB, nullptr, nullptr, out, VOCAB, DMODEL);
}
